// LegacyGriffinRecurrentPath_49890340110995
// MI455X (gfx1250) — matrix operand order repaired and checked by reading, not compiled, not run on hardware
//
#include <hip/hip_runtime.h>

// Problem constants (match reference)
#define B_ 4
#define S_ 4096
#define D_ 2048
#define NR_ 16384          // B_*S_ rows
#define CHUNKS_ 64         // scan chunks
#define CLEN_ 64           // scan chunk length (CHUNKS_*CLEN_ == S_)
#define KT_ (D_/32)        // 64 K-tiles of 32 for bf16 WMMA

typedef __bf16 bf16x16 __attribute__((ext_vector_type(16)));
typedef float  f32x8   __attribute__((ext_vector_type(8)));

__device__ __forceinline__ unsigned short f32_to_bf16(float f) {
    unsigned int u = __float_as_uint(f);
    u += 0x7fffu + ((u >> 16) & 1u);   // round-to-nearest-even
    return (unsigned short)(u >> 16);
}

__device__ __forceinline__ float sigmoidf_(float v) {
    return 1.0f / (1.0f + __expf(-v));
}

__device__ __forceinline__ f32x8 wmma_bf16(bf16x16 a, bf16x16 b, f32x8 c) {
    return __builtin_amdgcn_wmma_f32_16x16x32_bf16(false, a, false, b,
                                                   (short)0, c, false, false);
}

// ---------------------------------------------------------------------------
// Pack weight W[K=D_][N=D_] (row-major) into WMMA B-fragment tile order.
// Tile (kt, nt): 512 bf16 = 32 lanes x 16 elems.
// lane l, elem e:  k = kt*32 + (e>=8 ? 16 : 0) + (l>=16 ? 8 : 0) + (e&7) ; n = nt*16 + (l&15)
__global__ void pack_weight_kernel(const float* __restrict__ W,
                                   unsigned short* __restrict__ dst) {
    int t = blockIdx.x * 256 + threadIdx.x;          // D_*D_ threads
    int tile   = t >> 9;
    int within = t & 511;
    int lane = within >> 4;
    int e    = within & 15;
    int nt = tile % (D_ / 16);
    int kt = tile / (D_ / 16);
    int k = kt * 32 + ((e >> 3) << 4) + ((lane >> 4) << 3) + (e & 7);
    int n = nt * 16 + (lane & 15);
    dst[t] = f32_to_bf16(W[(size_t)k * D_ + n]);
}

// ---------------------------------------------------------------------------
// Fused projection GEMM: one 16x16 output tile per wave for each of
// inp = silu(x@W_in + b_in), dec = sigmoid(x@W_dec + b_dec + decay_bias),
// gate = sigmoid(x@W_gate + b_gate).
// 8 waves/block share a double-buffered 16x32 A tile (fp32 x converted to
// bf16 during LDS staging); B fragments are register-pipelined one K-step
// ahead so their L2 latency hides behind the 3 WMMAs.
__global__ void proj_gemm_kernel(const float* __restrict__ x,
                                 const unsigned short* __restrict__ WpkI,
                                 const unsigned short* __restrict__ WpkD,
                                 const unsigned short* __restrict__ WpkG,
                                 const float* __restrict__ biasI,
                                 const float* __restrict__ biasD,
                                 const float* __restrict__ biasG,
                                 const float* __restrict__ decay_bias,
                                 float* __restrict__ inp,
                                 float* __restrict__ dec,
                                 float* __restrict__ gate) {
    const int ntGroups = D_ / (16 * 8);              // 16 groups of 8 nt
    int mt     = blockIdx.x / ntGroups;
    int ntBase = (blockIdx.x % ntGroups) * 8;
    int wave = threadIdx.x >> 5;
    int lane = threadIdx.x & 31;
    int nt = ntBase + wave;
    int m0 = mt * 16;

    __shared__ __align__(32) unsigned short ldsA[2][512];

    // A staging map: thread -> (fragment lane, dword pair p)
    int tLane = threadIdx.x >> 3;                    // 0..31
    int p     = threadIdx.x & 7;                     // 0..7
    int am    = tLane & 15;                          // A row within tile
    int koff  = (tLane >> 4) * 8;                    // +8 K for upper lanes
    int ke    = (p < 4) ? (2 * p) : (2 * p + 8);     // K of first elem of pair
    const size_t aRow = (size_t)(m0 + am) * D_ + koff + ke;

    const size_t bOff    = (size_t)nt * 512 + lane * 16;
    const size_t bStride = (size_t)(D_ / 16) * 512;  // elems per K-tile slab

    f32x8 cI = {}; f32x8 cD = {}; f32x8 cG = {};

    // prologue: stage A(0), preload B(0)
    {
        float2 f = *(const float2*)&x[aRow];
        *(unsigned int*)&ldsA[0][tLane * 16 + 2 * p] =
            (unsigned)f32_to_bf16(f.x) | ((unsigned)f32_to_bf16(f.y) << 16);
    }
    bf16x16 nI = *(const bf16x16*)&WpkI[bOff];
    bf16x16 nD = *(const bf16x16*)&WpkD[bOff];
    bf16x16 nG = *(const bf16x16*)&WpkG[bOff];
    __syncthreads();

    for (int kt = 0; kt < KT_; ++kt) {
        int buf = kt & 1;
        bf16x16 fI = nI, fD = nD, fG = nG;
        if (kt + 1 < KT_) {
            // issue next iteration's loads before this iteration's WMMAs
            size_t nb = bOff + (size_t)(kt + 1) * bStride;
            nI = *(const bf16x16*)&WpkI[nb];
            nD = *(const bf16x16*)&WpkD[nb];
            nG = *(const bf16x16*)&WpkG[nb];
            float2 f = *(const float2*)&x[aRow + (size_t)(kt + 1) * 32];
            *(unsigned int*)&ldsA[buf ^ 1][tLane * 16 + 2 * p] =
                (unsigned)f32_to_bf16(f.x) | ((unsigned)f32_to_bf16(f.y) << 16);
        }
        if (kt + 2 < KT_) {
            size_t pb = bOff + (size_t)(kt + 2) * bStride;
            __builtin_prefetch(&WpkI[pb], 0, 1);
            __builtin_prefetch(&WpkD[pb], 0, 1);
            __builtin_prefetch(&WpkG[pb], 0, 1);
        }
        bf16x16 a = *(const bf16x16*)&ldsA[buf][lane * 16];
        cI = wmma_bf16(a, fI, cI);
        cD = wmma_bf16(a, fD, cD);
        cG = wmma_bf16(a, fG, cG);
        __syncthreads();
    }

    // epilogue: C/D layout row = r + 8*(lane>=16), col = lane&15
    int n     = nt * 16 + (lane & 15);
    int rbase = m0 + ((lane >> 4) << 3);
    float bI = biasI[n];
    float bD = biasD[n] + decay_bias[n];
    float bG = biasG[n];
#pragma unroll
    for (int r = 0; r < 8; ++r) {
        size_t o = (size_t)(rbase + r) * D_ + n;
        float vi = cI[r] + bI;
        inp[o]  = vi * sigmoidf_(vi);            // silu
        dec[o]  = sigmoidf_(cD[r] + bD);
        gate[o] = sigmoidf_(cG[r] + bG);
    }
}

// ---------------------------------------------------------------------------
// Scan pass 1: per chunk compute P = prod(d), Q = affine result from state 0.
// thread t = ((b*CHUNKS_ + c)*D_) + d
__global__ void scan_pass1_kernel(const float* __restrict__ dec,
                                  const float* __restrict__ inp,
                                  float* __restrict__ P, float* __restrict__ Q) {
    int t = blockIdx.x * 256 + threadIdx.x;
    int d  = t % D_;
    int bc = t / D_;
    int c  = bc % CHUNKS_;
    int b  = bc / CHUNKS_;
    float pp = 1.0f, qq = 0.0f;
    size_t base = ((size_t)b * S_ + (size_t)c * CLEN_) * D_ + d;
    for (int j = 0; j < CLEN_; ++j) {
        float dd = dec[base + (size_t)j * D_];
        float ii = inp[base + (size_t)j * D_];
        qq = dd * qq + (1.0f - dd) * ii;
        pp *= dd;
    }
    P[t] = pp;
    Q[t] = qq;
}

// Scan pass 2: sequential prefix over the 64 chunks (8192 threads).
// initState[(b*CHUNKS_+c)*D_+d] = state entering chunk c.
__global__ void scan_pass2_kernel(const float* __restrict__ P,
                                  const float* __restrict__ Q,
                                  float* __restrict__ initState) {
    int t = blockIdx.x * 256 + threadIdx.x;          // b*D_ + d
    int d = t % D_;
    int b = t / D_;
    float carry = 0.0f;
    for (int c = 0; c < CHUNKS_; ++c) {
        size_t idx = ((size_t)b * CHUNKS_ + c) * D_ + d;
        initState[idx] = carry;
        carry = P[idx] * carry + Q[idx];
    }
}

// Scan pass 3: replay each chunk from its initial state; y = gate*state (bf16)
__global__ void scan_pass3_kernel(const float* __restrict__ dec,
                                  const float* __restrict__ inp,
                                  const float* __restrict__ gate,
                                  const float* __restrict__ initState,
                                  unsigned short* __restrict__ yb) {
    int t = blockIdx.x * 256 + threadIdx.x;
    int d  = t % D_;
    int bc = t / D_;
    int c  = bc % CHUNKS_;
    int b  = bc / CHUNKS_;
    float state = initState[t];
    size_t base = ((size_t)b * S_ + (size_t)c * CLEN_) * D_ + d;
    for (int j = 0; j < CLEN_; ++j) {
        size_t o = base + (size_t)j * D_;
        float dd = dec[o];
        state = dd * state + (1.0f - dd) * inp[o];
        yb[o] = f32_to_bf16(gate[o] * state);
    }
}

// ---------------------------------------------------------------------------
// Output GEMM: out = y_bf16 @ W_out_bf16 + b_out (fp32 result).
// A tile (y, already bf16 in memory) is staged via the gfx1250 async
// global->LDS DMA path (ASYNCcnt), double-buffered; B register-pipelined.
__global__ void out_gemm_kernel(const unsigned short* __restrict__ yb,
                                const unsigned short* __restrict__ WpkO,
                                const float* __restrict__ biasO,
                                float* __restrict__ out) {
    const int ntGroups = D_ / (16 * 8);
    int mt     = blockIdx.x / ntGroups;
    int ntBase = (blockIdx.x % ntGroups) * 8;
    int wave = threadIdx.x >> 5;
    int lane = threadIdx.x & 31;
    int nt = ntBase + wave;
    int m0 = mt * 16;

    __shared__ __align__(32) unsigned short ldsA[2][512];

    int tLane = threadIdx.x >> 3;
    int p     = threadIdx.x & 7;
    int am    = tLane & 15;
    int koff  = (tLane >> 4) * 8;
    int ke    = (p < 4) ? (2 * p) : (2 * p + 8);
    const size_t aRow = (size_t)(m0 + am) * D_ + koff + ke;

    const size_t bOff    = (size_t)nt * 512 + lane * 16;
    const size_t bStride = (size_t)(D_ / 16) * 512;

    f32x8 cO = {};

    // async copy of this thread's 4 bytes of the A tile into LDS buffer
    auto stageA = [&](int kt, int buf) {
        unsigned ldsOff =
            (unsigned)(unsigned long long)&ldsA[buf][tLane * 16 + 2 * p];
        const unsigned short* g = &yb[aRow + (size_t)kt * 32];
        asm volatile("global_load_async_to_lds_b32 %0, %1, off"
                     :: "v"(ldsOff), "v"(g) : "memory");
    };

    stageA(0, 0);
    bf16x16 nO = *(const bf16x16*)&WpkO[bOff];
    asm volatile("s_wait_asynccnt 0x0" ::: "memory");
    __syncthreads();

    for (int kt = 0; kt < KT_; ++kt) {
        int buf = kt & 1;
        bf16x16 fO = nO;
        if (kt + 1 < KT_) {
            nO = *(const bf16x16*)&WpkO[bOff + (size_t)(kt + 1) * bStride];
            stageA(kt + 1, buf ^ 1);
        }
        if (kt + 2 < KT_) {
            __builtin_prefetch(&WpkO[bOff + (size_t)(kt + 2) * bStride], 0, 1);
        }
        bf16x16 a = *(const bf16x16*)&ldsA[buf][lane * 16];
        cO = wmma_bf16(a, fO, cO);
        // all of this wave's async LDS writes must land before the barrier
        asm volatile("s_wait_asynccnt 0x0" ::: "memory");
        __syncthreads();
    }

    int n     = nt * 16 + (lane & 15);
    int rbase = m0 + ((lane >> 4) << 3);
    float bO = biasO[n];
#pragma unroll
    for (int r = 0; r < 8; ++r) {
        out[(size_t)(rbase + r) * D_ + n] = cO[r] + bO;
    }
}

// ---------------------------------------------------------------------------
extern "C" void kernel_launch(void* const* d_in, const int* in_sizes, int n_in,
                              void* d_out, int out_size, void* d_ws, size_t ws_size,
                              hipStream_t stream) {
    const float* x          = (const float*)d_in[0];
    const float* W_in       = (const float*)d_in[1];
    const float* b_in       = (const float*)d_in[2];
    const float* W_decay    = (const float*)d_in[3];
    const float* b_decay    = (const float*)d_in[4];
    const float* W_gate     = (const float*)d_in[5];
    const float* b_gate     = (const float*)d_in[6];
    const float* W_out      = (const float*)d_in[7];
    const float* b_out      = (const float*)d_in[8];
    const float* decay_bias = (const float*)d_in[9];
    float* out = (float*)d_out;
    char*  ws  = (char*)d_ws;

    // workspace layout (all sizes multiples of 256B)
    const size_t szW    = (size_t)D_ * D_ * 2;           // one packed weight
    const size_t szF32  = (size_t)NR_ * D_ * 4;          // one fp32 intermediate
    const size_t szYB   = (size_t)NR_ * D_ * 2;          // y bf16
    const size_t szPQ   = (size_t)B_ * D_ * CHUNKS_ * 4; // per-chunk scalar

    size_t off = 0;
    unsigned short* WpkI  = (unsigned short*)(ws + off); off += szW;
    unsigned short* WpkD  = (unsigned short*)(ws + off); off += szW;
    unsigned short* WpkG  = (unsigned short*)(ws + off); off += szW;
    unsigned short* WpkO  = (unsigned short*)(ws + off); off += szW;
    float* inp            = (float*)(ws + off);          off += szF32;
    float* dec            = (float*)(ws + off);          off += szF32;
    float* gate           = (float*)(ws + off);          off += szF32;
    unsigned short* yb    = (unsigned short*)(ws + off); off += szYB;
    float* Pbuf           = (float*)(ws + off);          off += szPQ;
    float* Qbuf           = (float*)(ws + off);          off += szPQ;
    float* initState      = (float*)(ws + off);          off += szPQ;

    // 1) weight packing (fp32 -> bf16 WMMA fragment order)
    {
        int wb = (D_ * D_) / 256;
        pack_weight_kernel<<<dim3(wb), dim3(256), 0, stream>>>(W_in,    WpkI);
        pack_weight_kernel<<<dim3(wb), dim3(256), 0, stream>>>(W_decay, WpkD);
        pack_weight_kernel<<<dim3(wb), dim3(256), 0, stream>>>(W_gate,  WpkG);
        pack_weight_kernel<<<dim3(wb), dim3(256), 0, stream>>>(W_out,   WpkO);
    }

    // 2) fused projections (3 GEMMs + activations, x converted in-kernel)
    {
        int blocks = (NR_ / 16) * (D_ / (16 * 8));   // 1024 * 16
        proj_gemm_kernel<<<dim3(blocks), dim3(256), 0, stream>>>(
            x, WpkI, WpkD, WpkG, b_in, b_decay, b_gate, decay_bias,
            inp, dec, gate);
    }

    // 3) chunked affine scan (3 passes)
    {
        int t1 = B_ * D_ * CHUNKS_;                  // 524288
        scan_pass1_kernel<<<dim3(t1 / 256), dim3(256), 0, stream>>>(dec, inp, Pbuf, Qbuf);
        int t2 = B_ * D_;                            // 8192
        scan_pass2_kernel<<<dim3(t2 / 256), dim3(256), 0, stream>>>(Pbuf, Qbuf, initState);
        scan_pass3_kernel<<<dim3(t1 / 256), dim3(256), 0, stream>>>(dec, inp, gate, initState, yb);
    }

    // 4) output projection
    {
        int blocks = (NR_ / 16) * (D_ / (16 * 8));
        out_gemm_kernel<<<dim3(blocks), dim3(256), 0, stream>>>(yb, WpkO, b_out, out);
    }
}
